// QRNNLayer_8426725834998
// MI455X (gfx1250) — matrix operand order repaired and checked by reading, not compiled, not run on hardware
//
#include <hip/hip_runtime.h>
#include <hip/hip_bf16.h>

#define SEQ    2048
#define BATCH  16
#define DIM    1024
#define SB     (SEQ * BATCH)      // 32768 rows of the gate GEMM
#define CH     (BATCH * DIM)      // 16384 independent recurrence channels
#define NCHUNK 64
#define CLEN   (SEQ / NCHUNK)     // 32 steps per chunk

// ---- CDNA5 WMMA vector types (wave32, 16x16x32 bf16 -> f32) ----
typedef __bf16 bf16_t;
typedef __attribute__((ext_vector_type(16))) bf16_t v16bf;
typedef __attribute__((ext_vector_type(8)))  bf16_t v8bf;
typedef __attribute__((ext_vector_type(8)))  float  v8f;
typedef __attribute__((ext_vector_type(4)))  unsigned int uint4v;

union Frag {
    v16bf v;
    v8bf  h[2];
};

__device__ __forceinline__ unsigned short f2bf(float f) {
    union { float f; unsigned u; } x; x.f = f;
    unsigned r = x.u + 0x7FFFu + ((x.u >> 16) & 1u);   // round-to-nearest-even
    return (unsigned short)(r >> 16);
}

__device__ __forceinline__ float sigmoidf_(float x) {
    return 1.0f / (1.0f + __expf(-x));
}
__device__ __forceinline__ float tanhf_(float x) {
    return 2.0f / (1.0f + __expf(-2.0f * x)) - 1.0f;
}

// ------------------------------------------------------------------
// Kernel 0: fp32 -> bf16 conversion pass (bandwidth-bound; keeps all
// conversion VALU out of the WMMA inner loop). 8 elements / thread.
// ------------------------------------------------------------------
__global__ __launch_bounds__(256)
void cvt_f32_bf16(const float* __restrict__ src, unsigned short* __restrict__ dst)
{
    const size_t i = ((size_t)blockIdx.x * 256 + threadIdx.x) * 8;
    const float4 a = *reinterpret_cast<const float4*>(src + i);
    const float4 b = *reinterpret_cast<const float4*>(src + i + 4);
    uint4v u;
    u.x = (unsigned)f2bf(a.x) | ((unsigned)f2bf(a.y) << 16);
    u.y = (unsigned)f2bf(a.z) | ((unsigned)f2bf(a.w) << 16);
    u.z = (unsigned)f2bf(b.x) | ((unsigned)f2bf(b.y) << 16);
    u.w = (unsigned)f2bf(b.z) | ((unsigned)f2bf(b.w) << 16);
    *reinterpret_cast<uint4v*>(dst + i) = u;
}

// ------------------------------------------------------------------
// Kernel 1: fused gate GEMM + activations (bf16 WMMA, f32 accum).
// Block tile 128(M) x 64(d) x 3 gates, K-stage = 64, software-pipelined:
//   fetch(k+1) -> wmma(k) -> lds_write(k+1) -> barrier
// Epilogue:  Aarr = tanh(z)*sigmoid(f), Garr = 1-sigmoid(f),
//            Oarr = sigmoid(o) (parked in d_out H slot).
// ------------------------------------------------------------------
#define BM   128
#define BD   64
#define KS2  64          // K per LDS stage (2 WMMA sub-steps of 32)
#define LDA  72          // row stride (halves): 144 B -> 16B-aligned frags
#define NSTG (DIM / KS2) // 16 stages

__global__ __launch_bounds__(256)
void qrnn_gates_gemm(const unsigned short* __restrict__ Xb,
                     const unsigned short* __restrict__ Wb,
                     const float* __restrict__ bias,
                     float* __restrict__ Aarr,
                     float* __restrict__ Garr,
                     float* __restrict__ Oarr)
{
    __shared__ unsigned short lA[2][BM * LDA];        // 128 x 64 bf16 (+pad)
    __shared__ unsigned short lB[2][3 * BD * LDA];    // 192 x 64 bf16 (+pad)

    const int tid   = threadIdx.x;
    const int lane  = tid & 31;
    const int wave  = tid >> 5;       // 0..7
    const int wm    = wave >> 1;      // 0..3  (M sub-block)
    const int wd    = wave & 1;       // 0..1  (d sub-block)
    const int lhalf = (lane < 16) ? 0 : 1;
    const int l15   = lane & 15;

    const int mBase = blockIdx.x * BM;
    const int dBase = blockIdx.y * BD;

    v8f acc[2][3][2];
    #pragma unroll
    for (int mt = 0; mt < 2; mt++)
        #pragma unroll
        for (int g = 0; g < 3; g++)
            #pragma unroll
            for (int dt = 0; dt < 2; dt++)
                #pragma unroll
                for (int r = 0; r < 8; r++)
                    acc[mt][g][dt][r] = 0.0f;

    // Per-thread staging slots (16B chunks of 8 bf16):
    //  A: 128 rows x 8 chunks = 1024 chunks -> 4/thread
    //  B: 192 rows x 8 chunks = 1536 chunks -> 6/thread
    uint4v ra[4], rb[6];

    auto fetch = [&](int k0) {
        #pragma unroll
        for (int i = 0; i < 4; i++) {
            const int slot = tid * 4 + i;
            const int row = slot >> 3, c8 = slot & 7;
            const unsigned short* gp =
                Xb + (size_t)(mBase + row) * DIM + k0 + c8 * 8;
            ra[i] = *reinterpret_cast<const uint4v*>(gp);
            if (k0 + 2 * KS2 < DIM)                       // warm L2/HBM 2 stages out
                __builtin_prefetch(gp + 2 * KS2, 0, 0);   // -> global_prefetch_b8
        }
        #pragma unroll
        for (int i = 0; i < 6; i++) {
            const int slot = tid * 6 + i;
            const int row = slot >> 3, c8 = slot & 7;     // row 0..191
            const int gate = row >> 6, n = row & 63;
            rb[i] = *reinterpret_cast<const uint4v*>(
                Wb + (size_t)(gate * DIM + dBase + n) * DIM + k0 + c8 * 8);
        }
    };
    auto lds_write = [&](int buf) {
        #pragma unroll
        for (int i = 0; i < 4; i++) {
            const int slot = tid * 4 + i;
            const int row = slot >> 3, c8 = slot & 7;
            *reinterpret_cast<uint4v*>(&lA[buf][row * LDA + c8 * 8]) = ra[i];
        }
        #pragma unroll
        for (int i = 0; i < 6; i++) {
            const int slot = tid * 6 + i;
            const int row = slot >> 3, c8 = slot & 7;
            *reinterpret_cast<uint4v*>(&lB[buf][row * LDA + c8 * 8]) = rb[i];
        }
    };
    auto compute = [&](int buf) {
        #pragma unroll
        for (int ks = 0; ks < KS2; ks += 32) {
            // A frag (16x32): lanes 0-15 hold K {0..7,16..23}; 16-31 {8..15,24..31}
            Frag fa[2];
            #pragma unroll
            for (int mt = 0; mt < 2; mt++) {
                const unsigned short* p =
                    &lA[buf][(wm * 32 + mt * 16 + l15) * LDA + ks + lhalf * 8];
                fa[mt].h[0] = *reinterpret_cast<const v8bf*>(p);
                fa[mt].h[1] = *reinterpret_cast<const v8bf*>(p + 16);
            }
            // B frag (32x16): lane n=l15, lanes 0-15 hold K {0..7,16..23}; 16-31 {8..15,24..31}
            Frag fb[3][2];
            #pragma unroll
            for (int g = 0; g < 3; g++)
                #pragma unroll
                for (int dt = 0; dt < 2; dt++) {
                    const unsigned short* p =
                        &lB[buf][(g * 64 + wd * 32 + dt * 16 + l15) * LDA + ks + lhalf * 8];
                    fb[g][dt].h[0] = *reinterpret_cast<const v8bf*>(p);
                    fb[g][dt].h[1] = *reinterpret_cast<const v8bf*>(p + 16);
                }
            #pragma unroll
            for (int mt = 0; mt < 2; mt++)
                #pragma unroll
                for (int g = 0; g < 3; g++)
                    #pragma unroll
                    for (int dt = 0; dt < 2; dt++)
                        acc[mt][g][dt] = __builtin_amdgcn_wmma_f32_16x16x32_bf16(
                            false, fa[mt].v, false, fb[g][dt].v,
                            (short)0, acc[mt][g][dt], false, false);
        }
    };

    // Software pipeline over 16 K-stages.
    fetch(0);
    lds_write(0);
    int p = 0;
    for (int kt = 0; kt < NSTG; kt++) {
        __syncthreads();
        if (kt + 1 < NSTG) fetch((kt + 1) * KS2);   // global loads overlap wmma
        compute(p);
        if (kt + 1 < NSTG) lds_write(1 - p);
        p ^= 1;
    }

    // ---- fused epilogue: bias + activations + coefficient stores ----
    // C/D layout: lane l, reg r -> row = r + (l>=16 ? 8 : 0), col = l%16
    #pragma unroll
    for (int mt = 0; mt < 2; mt++) {
        const int mRow = mBase + wm * 32 + mt * 16 + lhalf * 8;
        #pragma unroll
        for (int dt = 0; dt < 2; dt++) {
            const int d  = dBase + wd * 32 + dt * 16 + l15;
            const float bz = bias[d];
            const float bf = bias[DIM + d];
            const float bo = bias[2 * DIM + d];
            #pragma unroll
            for (int r = 0; r < 8; r++) {
                const size_t idx = (size_t)(mRow + r) * DIM + d;
                const float z = tanhf_(acc[mt][0][dt][r] + bz);
                const float f = sigmoidf_(acc[mt][1][dt][r] + bf);
                const float o = sigmoidf_(acc[mt][2][dt][r] + bo);
                Aarr[idx] = z * f;
                Garr[idx] = 1.0f - f;
                Oarr[idx] = o;          // parked in d_out; consumed by pass 3
            }
        }
    }
}

// ------------------------------------------------------------------
// Kernel 2a: compose each 32-step chunk of the linear recurrence
//   c' = a + g*c  ==>  (A_chunk, G_chunk) with c_out = A + G*c_in
// ------------------------------------------------------------------
__global__ __launch_bounds__(256)
void qrnn_chunk_reduce(const float* __restrict__ Aarr,
                       const float* __restrict__ Garr,
                       float* __restrict__ chunkA,
                       float* __restrict__ chunkG)
{
    const int ch    = blockIdx.x * 256 + threadIdx.x;
    const int chunk = blockIdx.y;
    float Ac = 0.0f, Gc = 1.0f;
    size_t base = (size_t)chunk * CLEN * CH + ch;
    #pragma unroll 4
    for (int s = 0; s < CLEN; s++) {
        const float a = Aarr[base], g = Garr[base];
        Ac = a + g * Ac;
        Gc = g * Gc;
        base += CH;
    }
    chunkA[(size_t)chunk * CH + ch] = Ac;
    chunkG[(size_t)chunk * CH + ch] = Gc;
}

// ------------------------------------------------------------------
// Kernel 2b: serial scan over the 64 chunk summaries (tiny) ->
// per-chunk starting hidden state + final C written to d_out tail.
// ------------------------------------------------------------------
__global__ __launch_bounds__(256)
void qrnn_chunk_scan(const float* __restrict__ chunkA,
                     const float* __restrict__ chunkG,
                     const float* __restrict__ hidden,
                     float* __restrict__ chunkC,
                     float* __restrict__ outTail)
{
    const int ch = blockIdx.x * 256 + threadIdx.x;
    float c = hidden[ch];
    #pragma unroll 4
    for (int k = 0; k < NCHUNK; k++) {
        chunkC[(size_t)k * CH + ch] = c;
        c = chunkA[(size_t)k * CH + ch] + chunkG[(size_t)k * CH + ch] * c;
    }
    outTail[ch] = c;   // C[-1]
}

// ------------------------------------------------------------------
// Kernel 2c: replay each chunk with known starting state;
// H = sigmoid(O) * C, with sigmoid(O) already in d_out (in place).
// ------------------------------------------------------------------
__global__ __launch_bounds__(256)
void qrnn_chunk_apply(const float* __restrict__ Aarr,
                      const float* __restrict__ Garr,
                      const float* __restrict__ chunkC,
                      float* __restrict__ H)
{
    const int ch    = blockIdx.x * 256 + threadIdx.x;
    const int chunk = blockIdx.y;
    float c = chunkC[(size_t)chunk * CH + ch];
    size_t base = (size_t)chunk * CLEN * CH + ch;
    #pragma unroll 4
    for (int s = 0; s < CLEN; s++) {
        const float a = Aarr[base], g = Garr[base];
        c = a + g * c;
        H[base] = H[base] * c;
        base += CH;
    }
}

extern "C" void kernel_launch(void* const* d_in, const int* in_sizes, int n_in,
                              void* d_out, int out_size, void* d_ws, size_t ws_size,
                              hipStream_t stream) {
    (void)in_sizes; (void)n_in; (void)out_size; (void)ws_size;
    const float* X      = (const float*)d_in[0];   // (S,B,D)
    const float* hidden = (const float*)d_in[1];   // (B,D)
    const float* W      = (const float*)d_in[2];   // (3D,D)
    const float* b      = (const float*)d_in[3];   // (3D,)

    float* H     = (float*)d_out;                  // (S,B,D)
    float* Ctail = H + (size_t)SEQ * CH;           // (1,B,D)

    float* ws     = (float*)d_ws;
    float* Aarr   = ws;                                  // S*B*D f32
    float* Garr   = Aarr   + (size_t)SEQ * CH;           // S*B*D f32
    float* chunkA = Garr   + (size_t)SEQ * CH;           // NCHUNK*CH f32
    float* chunkG = chunkA + (size_t)NCHUNK * CH;
    float* chunkC = chunkG + (size_t)NCHUNK * CH;
    unsigned short* Xb = (unsigned short*)(chunkC + (size_t)NCHUNK * CH); // S*B*D bf16
    unsigned short* Wb = Xb + (size_t)SEQ * CH;                           // 3*D*D bf16

    // fp32 -> bf16 pre-passes (2048 elems per block)
    cvt_f32_bf16<<<(SEQ * (size_t)CH) / 2048, 256, 0, stream>>>(X, Xb);
    cvt_f32_bf16<<<(3 * DIM * DIM) / 2048, 256, 0, stream>>>(W, Wb);

    dim3 gGemm(SB / BM, DIM / BD);      // 256 x 16 workgroups
    qrnn_gates_gemm<<<gGemm, 256, 0, stream>>>(Xb, Wb, b, Aarr, Garr, H);

    dim3 gChunk(CH / 256, NCHUNK);      // 64 x 64
    qrnn_chunk_reduce<<<gChunk, 256, 0, stream>>>(Aarr, Garr, chunkA, chunkG);
    qrnn_chunk_scan<<<CH / 256, 256, 0, stream>>>(chunkA, chunkG, hidden, chunkC, Ctail);
    qrnn_chunk_apply<<<gChunk, 256, 0, stream>>>(Aarr, Garr, chunkC, H);
}
